// CnnLstmPolicy_20057497272809
// MI455X (gfx1250) — compile-verified, matrix operand order checked by reading, not run on hardware
//
#include <hip/hip_runtime.h>
#include <stdint.h>

// ---------------------------------------------------------------------------
// CnnLstmPolicy forward for MI455X (gfx1250), wave32 + WMMA f16->f32.
//
//   1. fp32->fp16 weight conversion; conv weights reordered to k = d*C + c
//      (tap-major, channel-minor) so conv A-fragments are contiguous v8h LDS
//      loads; b_ih+b_hh fused; s1_w K-padded 452->480
//   2. scalar extraction  obs[:, :452] -> f16 padded (K=480)
//   3. scalar MLP: two WMMA GEMMs -> emb[:,128:192]
//   4. per-sample fused CNN: grid staged into LDS via async-to-LDS copies
//      (GLOBAL_LOAD_ASYNC_TO_LDS_B32 + s_wait_asynccnt, guarded by
//      __has_builtin), conv1 VALU f32, conv2/conv3 implicit-GEMM WMMA with
//      channel-last LDS activations (ds_load_b128 fragments), chunked x4,
//      + FC WMMA GEMM -> emb[:,0:128]
//   5. xg = emb @ w_ih^T + (b_ih+b_hh)  (WMMA, f16 out, L2-resident 33MB)
//   6. sequential LSTM: 1 workgroup, 32 waves, w_hh B-fragments pinned in
//      VGPRs across all 256 steps; h/c/gates in 224KB LDS; next-step xg
//      panel prefetched (global_prefetch) during compute
//   7. value head post-pass (hoisted off the serial path)
//   8. policy head WMMA GEMM (4 N-subtiles/wave) -> fp32 logits
//
// Requires ws_size >= ~140 MB.
// ---------------------------------------------------------------------------

typedef _Float16 half_t;
typedef __attribute__((ext_vector_type(16))) _Float16 v16h;
typedef __attribute__((ext_vector_type(8)))  _Float16 v8h;
typedef __attribute__((ext_vector_type(8)))  float    v8f;

#if __has_builtin(__builtin_amdgcn_global_load_async_to_lds_b32) && \
    __has_builtin(__builtin_amdgcn_s_wait_asynccnt)
#define HAVE_ASYNC_LDS 1
// builtin params are int pointers: AS1 ("__device__") src, AS3 dst
typedef __attribute__((address_space(1))) int gint_t;
typedef __attribute__((address_space(3))) int lint_t;
#else
#define HAVE_ASYNC_LDS 0
#endif

#define T_STEPS   256
#define BATCH     128
#define NS        (T_STEPS * BATCH)      // 32768 samples
#define OBS_DIM   1604
#define SCALAR_DIM 452
#define SC_PAD    480
#define GRID_OFF  452
#define HID       128
#define GATES     512
#define EMB       192
#define N_ACT     2305
#define CNN_FLAT  2560
#define CHUNK     8192

// padded LDS activation dims (channel-last: [H][W][C])
#define P1H 34
#define P1W 20
#define P2H 18
#define P2W 11

// ----------------------------- WMMA helpers --------------------------------

__device__ __forceinline__ v16h cat8(v8h lo, v8h hi) {
  v16h r;
#pragma unroll
  for (int i = 0; i < 8; ++i) { r[i] = lo[i]; r[8 + i] = hi[i]; }
  return r;
}

__device__ __forceinline__ v8f wmma_f16(v16h a, v16h b, v8f c) {
  return __builtin_amdgcn_wmma_f32_16x16x32_f16(false, a, false, b,
                                                (short)0, c, false, false);
}

__device__ __forceinline__ float sigf(float x) {
  return 1.f / (1.f + __expf(-x));
}

// --------------------------- small utility kernels -------------------------

__global__ void cvt_f16_kernel(const float* __restrict__ s,
                               half_t* __restrict__ d, int n) {
  int i = blockIdx.x * blockDim.x + threadIdx.x;
  if (i < n) d[i] = (half_t)s[i];
}

__global__ void cvt_pad_kernel(const float* __restrict__ s,
                               half_t* __restrict__ d,
                               int rows, int sk, int dk) {
  int i = blockIdx.x * blockDim.x + threadIdx.x;
  if (i >= rows * dk) return;
  int r = i / dk, k = i % dk;
  d[i] = (k < sk) ? (half_t)s[r * sk + k] : (half_t)0.f;
}

// conv weight reorder: OIHW (n,c,d) -> WMMA-B rows with k = d*C + c
__global__ void reorder_convw_kernel(const float* __restrict__ s,
                                     half_t* __restrict__ d, int N, int C) {
  int i = blockIdx.x * blockDim.x + threadIdx.x;
  if (i >= N * C * 9) return;
  int n = i / (C * 9), r = i % (C * 9), c = r / 9, dd = r % 9;
  d[(size_t)n * (C * 9) + dd * C + c] = (half_t)s[i];
}

__global__ void addvec_kernel(const float* __restrict__ a,
                              const float* __restrict__ b,
                              float* __restrict__ o, int n) {
  int i = blockIdx.x * blockDim.x + threadIdx.x;
  if (i < n) o[i] = a[i] + b[i];
}

__global__ void extract_scalars_kernel(const float* __restrict__ obs,
                                       half_t* __restrict__ dst) {
  int i = blockIdx.x * blockDim.x + threadIdx.x;
  if (i >= NS * SC_PAD) return;
  int r = i / SC_PAD, k = i % SC_PAD;
  dst[i] = (k < SCALAR_DIM) ? (half_t)obs[(size_t)r * OBS_DIM + k]
                            : (half_t)0.f;
}

// value head post-pass: values[s] = outs[s,:] . val_w + val_b  (K=128)
__global__ void value_head_kernel(const half_t* __restrict__ outs,
                                  const float* __restrict__ valw,
                                  const float* __restrict__ valb,
                                  float* __restrict__ values) {
  int s = blockIdx.x * blockDim.x + threadIdx.x;
  if (s >= NS) return;
  const half_t* row = outs + (size_t)s * HID;
  float acc = valb[0];
#pragma unroll
  for (int j = 0; j < HID; ++j) acc += (float)row[j] * valw[j];
  values[s] = acc;
}

// ------------------------------ generic GEMM -------------------------------
// C[M,N] = act(A[M,K]f16 * Bt[N,K]f16^T + bias). One 16-row M tile x 4 N
// subtiles per wave (one A fragment feeds 4 WMMAs). Fragment layout per
// CDNA5 ISA: lane<16 holds K {k0..k0+7, k0+16..k0+23}, lane>=16 holds
// {k0+8.., k0+24..} -> two v8h (global_load_b128). M,K multiples of 16/32.

template <bool RELU, bool F16OUT>
__global__ void __launch_bounds__(256)
gemm_wmma_kernel(const half_t* __restrict__ A, int lda,
                 const half_t* __restrict__ Bt, int ldb,
                 const float* __restrict__ bias,
                 void* __restrict__ Cout, int ldc,
                 int M, int N, int K) {
  const int lane = threadIdx.x & 31;
  const int wid  = blockIdx.x * 8 + (threadIdx.x >> 5);
  const int tM  = M >> 4;
  const int tN  = (N + 15) >> 4;
  const int tNg = (tN + 3) >> 2;
  if (wid >= tM * tNg) return;             // wave-uniform
  const int mt = wid / tNg, ng = wid % tNg;
  const int row  = mt * 16 + (lane & 15);
  const int nb   = ng * 64 + (lane & 15);
  const int hsel = (lane & 16) ? 8 : 0;

  const half_t* arow = A + (size_t)row * lda;
  const half_t* bp[4];
#pragma unroll
  for (int s = 0; s < 4; ++s) bp[s] = Bt + (size_t)(nb + 16 * s) * ldb;

  v8f acc[4] = {{}, {}, {}, {}};
  for (int k0 = 0; k0 < K; k0 += 32) {
    const int kh = k0 + hsel;
    if (k0 + 32 < K) __builtin_prefetch(arow + kh + 32, 0, 0);
    v16h a = cat8(*(const v8h*)(arow + kh), *(const v8h*)(arow + kh + 16));
#pragma unroll
    for (int s = 0; s < 4; ++s) {
      v16h b = cat8(*(const v8h*)(bp[s] + kh), *(const v8h*)(bp[s] + kh + 16));
      acc[s] = wmma_f16(a, b, acc[s]);
    }
  }

#pragma unroll
  for (int s = 0; s < 4; ++s) {
    int ncol = ng * 64 + s * 16 + (lane & 15);
    if (ncol < N) {
      float bb = bias[ncol];
#pragma unroll
      for (int r = 0; r < 8; ++r) {
        int orow = mt * 16 + r + ((lane & 16) ? 8 : 0);
        float v = acc[s][r] + bb;
        if (RELU) v = fmaxf(v, 0.f);
        if (F16OUT) ((half_t*)Cout)[(size_t)orow * ldc + ncol] = (half_t)v;
        else        ((float*)Cout)[(size_t)orow * ldc + ncol]  = v;
      }
    }
  }
}

// --------------------------- fused per-sample CNN --------------------------
// 1 workgroup (256 thr = 8 waves) per sample. Dynamic LDS = 74304 B:
//   inp f32 [2][34][20] | b1 f16 [34][20][32] ch-last | b2 f16 [18][11][64]
// Grid staged with async-to-LDS B32 copies (waits with s_wait_asynccnt).
// conv1 is direct VALU f32; conv2/conv3 are implicit-GEMM WMMA with
// k = d*C + c so every A fragment is two ds_load_b128 with zero index math.

__global__ void __launch_bounds__(256)
cnn_kernel(const float* __restrict__ obs,
           const float* __restrict__ c1w, const float* __restrict__ c1b,
           const half_t* __restrict__ c2wr, const float* __restrict__ c2b,
           const half_t* __restrict__ c3wr, const float* __restrict__ c3b,
           half_t* __restrict__ conv3buf, int sample_base) {
  extern __shared__ char dynsmem[];
  float*  inpf = (float*)dynsmem;                    // [2][34][20] f32
  half_t* b1   = (half_t*)(dynsmem + 2 * P1H * P1W * 4); // [34][20][32]
  half_t* b2   = b1 + P1H * P1W * 32;                // [18][11][64]
  const int tid  = threadIdx.x;
  const int lane = tid & 31;
  const int wid  = tid >> 5;
  const int hsel = (lane & 16) ? 8 : 0;
  const int s = sample_base + blockIdx.x;

  for (int i = tid; i < 2 * P1H * P1W; i += 256) inpf[i] = 0.f;
  for (int i = tid; i < P1H * P1W * 32 + P2H * P2W * 64; i += 256)
    b1[i] = (half_t)0.f;
  __syncthreads();

  // stage grid interior into padded LDS
  const float* g = obs + (size_t)s * OBS_DIM + GRID_OFF;
#if HAVE_ASYNC_LDS
  for (int i = tid; i < 2 * 32 * 18; i += 256) {
    int c = i / 576, r = i % 576, y = r / 18, x = r % 18;
    float* lp = inpf + (c * P1H + y + 1) * P1W + x + 1;
    __builtin_amdgcn_global_load_async_to_lds_b32(
        (gint_t*)(g + i), (lint_t*)lp, 0, 0);
  }
  __builtin_amdgcn_s_wait_asynccnt(0);
#else
  for (int i = tid; i < 2 * 32 * 18; i += 256) {
    int c = i / 576, r = i % 576, y = r / 18, x = r % 18;
    inpf[(c * P1H + y + 1) * P1W + x + 1] = g[i];
  }
#endif
  __syncthreads();

  // conv1: 2->32 ch, 3x3 s1 p1, relu; write channel-last f16
  for (int i = tid; i < 32 * 32 * 18; i += 256) {
    int o = i / 576, r = i % 576, y = r / 18, x = r % 18;
    float acc = c1b[o];
#pragma unroll
    for (int c = 0; c < 2; ++c)
#pragma unroll
      for (int dy = 0; dy < 3; ++dy)
#pragma unroll
        for (int dx = 0; dx < 3; ++dx)
          acc += inpf[(c * P1H + y + dy) * P1W + x + dx] *
                 c1w[((o * 2 + c) * 3 + dy) * 3 + dx];
    b1[((y + 1) * P1W + (x + 1)) * 32 + o] = (half_t)fmaxf(acc, 0.f);
  }
  __syncthreads();

  // conv2 implicit GEMM: M=144 (16x9), N=64, K=288 (k = ks*32 + c)
  for (int tile = wid; tile < 9 * 4; tile += 8) {
    int mt = tile >> 2, nt = tile & 3;
    int p   = mt * 16 + (lane & 15);         // 0..143
    int oy2 = (p / 9) * 2, ox2 = (p % 9) * 2;
    int n   = nt * 16 + (lane & 15);
    const half_t* brow = c2wr + (size_t)n * 288;
    v8f acc = {};
#pragma unroll
    for (int ks = 0; ks < 9; ++ks) {
      const int dy = ks / 3, dx = ks % 3;    // compile-time after unroll
      const half_t* ap = b1 + ((oy2 + dy) * P1W + ox2 + dx) * 32 + hsel;
      v16h a = cat8(*(const v8h*)ap, *(const v8h*)(ap + 16));
      const half_t* bpk = brow + ks * 32 + hsel;
      v16h b = cat8(*(const v8h*)bpk, *(const v8h*)(bpk + 16));
      acc = wmma_f16(a, b, acc);
    }
    float bb = c2b[n];
#pragma unroll
    for (int r = 0; r < 8; ++r) {
      int pr = mt * 16 + r + ((lane & 16) ? 8 : 0);
      int ry = pr / 9, rx = pr % 9;
      b2[((ry + 1) * P2W + (rx + 1)) * 64 + n] =
          (half_t)fmaxf(acc[r] + bb, 0.f);
    }
  }
  __syncthreads();

  // conv3 implicit GEMM: M=40 (8x5, padded to 48), N=64, K=576 (k = d*64+c).
  // Rows p>=40 compute garbage from a clamped address but are never stored.
  for (int tile = wid; tile < 3 * 4; tile += 8) {
    int mt = tile >> 2, nt = tile & 3;
    int p  = mt * 16 + (lane & 15);
    int vp = (p < 40) ? p : 0;
    int oy2 = (vp / 5) * 2, ox2 = (vp % 5) * 2;
    int n  = nt * 16 + (lane & 15);
    const half_t* brow = c3wr + (size_t)n * 576;
    v8f acc = {};
#pragma unroll
    for (int ks = 0; ks < 18; ++ks) {
      const int d  = ks >> 1;
      const int c0 = (ks & 1) * 32;
      const int dy = d / 3, dx = d % 3;
      const half_t* ap =
          b2 + ((oy2 + dy) * P2W + ox2 + dx) * 64 + c0 + hsel;
      v16h a = cat8(*(const v8h*)ap, *(const v8h*)(ap + 16));
      const half_t* bpk = brow + d * 64 + c0 + hsel;
      v16h b = cat8(*(const v8h*)bpk, *(const v8h*)(bpk + 16));
      acc = wmma_f16(a, b, acc);
    }
    float bb = c3b[n];
    half_t* dst = conv3buf + (size_t)blockIdx.x * CNN_FLAT;
#pragma unroll
    for (int r = 0; r < 8; ++r) {
      int pr = mt * 16 + r + ((lane & 16) ? 8 : 0);
      if (pr < 40)                           // flat idx = o*40 + oy*5 + ox
        dst[n * 40 + pr] = (half_t)fmaxf(acc[r] + bb, 0.f);
    }
  }
}

// ------------------------------- LSTM scan ---------------------------------
// Single workgroup, 1024 threads = 32 waves. Wave w owns gate columns
// [16w,16w+16); its w_hh B-fragments (4 k-steps, 32 VGPRs) stay resident for
// all 256 steps. LDS (229376 B): gates f16[128][512] | h f16[128][128] |
// c f32[128][128]. xg is f16 (L2-resident); each thread prefetches one 128B
// block of the next step's xg panel while this step computes.

__global__ void __launch_bounds__(1024)
lstm_kernel(const half_t* __restrict__ xg, const unsigned char* __restrict__ done,
            const float* __restrict__ h0, const float* __restrict__ c0,
            const half_t* __restrict__ whh,
            half_t* __restrict__ outs,
            float* __restrict__ hf, float* __restrict__ cf) {
  extern __shared__ char dynsmem[];
  half_t* gS = (half_t*)dynsmem;                 // [128][512]
  half_t* hS = gS + 128 * 512;                   // [128][128]
  float*  cS = (float*)(hS + 128 * 128);         // [128][128]
  const int tid  = threadIdx.x;
  const int lane = tid & 31;
  const int wid  = tid >> 5;
  const int hsel = (lane & 16) ? 8 : 0;
  const int ncol = wid * 16 + (lane & 15);

  for (int e = tid; e < 128 * 128; e += 1024) {
    hS[e] = (half_t)h0[e];
    cS[e] = c0[e];
  }

  v16h bfr[4];
#pragma unroll
  for (int ks = 0; ks < 4; ++ks) {
    int kh = ks * 32 + hsel;
    bfr[ks] = cat8(*(const v8h*)(whh + (size_t)ncol * 128 + kh),
                   *(const v8h*)(whh + (size_t)ncol * 128 + kh + 16));
  }
  __syncthreads();

  for (int t = 0; t < T_STEPS; ++t) {
    if (t > 0) {  // h,c *= keep  (keep[t] = !done[t-1])
      for (int e = tid; e < 128 * 128; e += 1024) {
        int b = e >> 7;
        if (done[(size_t)(t - 1) * 128 + b]) {
          hS[e] = (half_t)0.f;
          cS[e] = 0.f;
        }
      }
    }
    __syncthreads();

    if (t + 1 < T_STEPS)  // pull next step's gate panel toward the WGP
      __builtin_prefetch(xg + (size_t)(t + 1) * 128 * 512 + tid * 64, 0, 0);

    const half_t* xgt = xg + (size_t)t * 128 * 512;
#pragma unroll
    for (int mt = 0; mt < 8; ++mt) {
      v8f acc = {};
#pragma unroll
      for (int ks = 0; ks < 4; ++ks) {
        int row = mt * 16 + (lane & 15);
        int kh  = ks * 32 + hsel;
        acc = wmma_f16(cat8(*(const v8h*)(hS + row * 128 + kh),
                            *(const v8h*)(hS + row * 128 + kh + 16)),
                       bfr[ks], acc);
      }
#pragma unroll
      for (int r = 0; r < 8; ++r) {
        int orow = mt * 16 + r + ((lane & 16) ? 8 : 0);
        gS[orow * 512 + ncol] =
            (half_t)(acc[r] + (float)xgt[(size_t)orow * 512 + ncol]);
      }
    }
    __syncthreads();

    for (int e = tid; e < 128 * 128; e += 1024) {
      int b = e >> 7, j = e & 127;
      float ig = sigf((float)gS[b * 512 + j]);
      float fg = sigf((float)gS[b * 512 + 128 + j]);
      float gg = tanhf((float)gS[b * 512 + 256 + j]);
      float og = sigf((float)gS[b * 512 + 384 + j]);
      float cn = fg * cS[e] + ig * gg;
      float hn = og * tanhf(cn);
      cS[e] = cn;
      hS[e] = (half_t)hn;
      outs[((size_t)t * 128 + b) * 128 + j] = (half_t)hn;
    }
    __syncthreads();
  }

  for (int e = tid; e < 128 * 128; e += 1024) {
    hf[e] = (float)hS[e];
    cf[e] = cS[e];
  }
}

// ------------------------------- host driver -------------------------------

static inline int gemm_blocks(int M, int N) {
  long tN  = (N + 15) >> 4;
  long tiles = (long)(M >> 4) * ((tN + 3) >> 2);
  return (int)((tiles + 7) / 8);
}

extern "C" void kernel_launch(void* const* d_in, const int* in_sizes, int n_in,
                              void* d_out, int out_size, void* d_ws,
                              size_t ws_size, hipStream_t stream) {
  (void)in_sizes; (void)n_in; (void)out_size; (void)ws_size;

  const float* obs  = (const float*)d_in[0];
  const float* h0   = (const float*)d_in[1];
  const float* c0   = (const float*)d_in[2];
  const unsigned char* done = (const unsigned char*)d_in[3]; // jnp bool_ = 1B
  const float* c1w  = (const float*)d_in[4];
  const float* c1b  = (const float*)d_in[5];
  const float* c2w  = (const float*)d_in[6];
  const float* c2b  = (const float*)d_in[7];
  const float* c3w  = (const float*)d_in[8];
  const float* c3b  = (const float*)d_in[9];
  const float* fcw  = (const float*)d_in[10];
  const float* fcb  = (const float*)d_in[11];
  const float* s1w  = (const float*)d_in[12];
  const float* s1b  = (const float*)d_in[13];
  const float* s2w  = (const float*)d_in[14];
  const float* s2b  = (const float*)d_in[15];
  const float* wih  = (const float*)d_in[16];
  const float* whh  = (const float*)d_in[17];
  const float* bih  = (const float*)d_in[18];
  const float* bhh  = (const float*)d_in[19];
  const float* polw = (const float*)d_in[20];
  const float* polb = (const float*)d_in[21];
  const float* valw = (const float*)d_in[22];
  const float* valb = (const float*)d_in[23];

  float* logits = (float*)d_out;                      // 32768 x 2305
  float* values = logits + (size_t)NS * N_ACT;
  float* hfp    = values + NS;
  float* cfp    = hfp + BATCH * HID;

  char* ws = (char*)d_ws;
  size_t off = 0;
  auto alloc = [&](size_t bytes) -> char* {
    char* p = ws + off;
    off = (off + bytes + 255) & ~(size_t)255;
    return p;
  };
  half_t* c2wr    = (half_t*)alloc(18432 * 2);
  half_t* c3wr    = (half_t*)alloc(36864 * 2);
  half_t* fcw16   = (half_t*)alloc((size_t)128 * CNN_FLAT * 2);
  half_t* s1w16   = (half_t*)alloc((size_t)64 * SC_PAD * 2);
  half_t* s2w16   = (half_t*)alloc(4096 * 2);
  half_t* wih16   = (half_t*)alloc((size_t)GATES * EMB * 2);
  half_t* whh16   = (half_t*)alloc((size_t)GATES * HID * 2);
  half_t* polw16  = (half_t*)alloc((size_t)N_ACT * HID * 2);
  float*  biassum = (float*)alloc(GATES * 4);
  half_t* sc_in   = (half_t*)alloc((size_t)NS * SC_PAD * 2);
  half_t* scbuf   = (half_t*)alloc((size_t)NS * 64 * 2);
  half_t* emb     = (half_t*)alloc((size_t)NS * EMB * 2);
  half_t* conv3buf= (half_t*)alloc((size_t)CHUNK * CNN_FLAT * 2);
  half_t* xg      = (half_t*)alloc((size_t)NS * GATES * 2);
  half_t* outs16  = (half_t*)alloc((size_t)NS * HID * 2);

  const int CNN_LDS =
      2 * P1H * P1W * 4 + (P1H * P1W * 32 + P2H * P2W * 64) * 2;  // 74304
  const int LSTM_LDS = 128 * 512 * 2 + 128 * 128 * 2 + 128 * 128 * 4;
  (void)hipFuncSetAttribute((const void*)cnn_kernel,
                            hipFuncAttributeMaxDynamicSharedMemorySize,
                            CNN_LDS);
  (void)hipFuncSetAttribute((const void*)lstm_kernel,
                            hipFuncAttributeMaxDynamicSharedMemorySize,
                            LSTM_LDS);

  auto cvt = [&](const float* s, half_t* d, int n) {
    cvt_f16_kernel<<<(n + 255) / 256, 256, 0, stream>>>(s, d, n);
  };

  // 1. weight conversion / reorder
  reorder_convw_kernel<<<(18432 + 255) / 256, 256, 0, stream>>>(c2w, c2wr, 64, 32);
  reorder_convw_kernel<<<(36864 + 255) / 256, 256, 0, stream>>>(c3w, c3wr, 64, 64);
  cvt(fcw, fcw16, 128 * CNN_FLAT);
  cvt(s2w, s2w16, 4096);
  cvt(wih, wih16, GATES * EMB);
  cvt(whh, whh16, GATES * HID);
  cvt(polw, polw16, N_ACT * HID);
  cvt_pad_kernel<<<(64 * SC_PAD + 255) / 256, 256, 0, stream>>>(
      s1w, s1w16, 64, SCALAR_DIM, SC_PAD);
  addvec_kernel<<<2, 256, 0, stream>>>(bih, bhh, biassum, GATES);

  // 2. scalar extraction
  extract_scalars_kernel<<<(NS * SC_PAD + 255) / 256, 256, 0, stream>>>(
      obs, sc_in);

  // 3. scalar MLP -> emb[:,128:192]
  gemm_wmma_kernel<true, true><<<gemm_blocks(NS, 64), 256, 0, stream>>>(
      sc_in, SC_PAD, s1w16, SC_PAD, s1b, scbuf, 64, NS, 64, SC_PAD);
  gemm_wmma_kernel<true, true><<<gemm_blocks(NS, 64), 256, 0, stream>>>(
      scbuf, 64, s2w16, 64, s2b, emb + 128, EMB, NS, 64, 64);

  // 4. fused CNN (chunked) + FC -> emb[:,0:128]
  for (int ch = 0; ch < NS / CHUNK; ++ch) {
    cnn_kernel<<<CHUNK, 256, CNN_LDS, stream>>>(
        obs, c1w, c1b, c2wr, c2b, c3wr, c3b, conv3buf, ch * CHUNK);
    gemm_wmma_kernel<true, true><<<gemm_blocks(CHUNK, 128), 256, 0, stream>>>(
        conv3buf, CNN_FLAT, fcw16, CNN_FLAT, fcb,
        emb + (size_t)ch * CHUNK * EMB, EMB, CHUNK, 128, CNN_FLAT);
  }

  // 5. xg = emb @ w_ih^T + (b_ih + b_hh)   (f16 out, L2-resident)
  gemm_wmma_kernel<false, true><<<gemm_blocks(NS, GATES), 256, 0, stream>>>(
      emb, EMB, wih16, EMB, biassum, xg, GATES, NS, GATES, EMB);

  // 6. sequential LSTM scan (h_f/c_f written at the end)
  lstm_kernel<<<1, 1024, LSTM_LDS, stream>>>(
      xg, done, h0, c0, whh16, outs16, hfp, cfp);

  // 7. value head (hoisted off the serial path)
  value_head_kernel<<<(NS + 255) / 256, 256, 0, stream>>>(
      outs16, valw, valb, values);

  // 8. policy head -> fp32 logits
  gemm_wmma_kernel<false, false><<<gemm_blocks(NS, N_ACT), 256, 0, stream>>>(
      outs16, HID, polw16, HID, polb, logits, N_ACT, NS, N_ACT, HID);
}
